// FlorenceDecoderAttention_1460288881463
// MI455X (gfx1250) — matrix operand order repaired and checked by reading, not compiled, not run on hardware
//
#include <hip/hip_runtime.h>

typedef __attribute__((ext_vector_type(16))) __bf16 v16bf;
typedef __attribute__((ext_vector_type(8)))  float  v8f;
typedef __attribute__((ext_vector_type(4)))  unsigned int v4u;
typedef __attribute__((ext_vector_type(8)))  int  v8i;
typedef __attribute__((ext_vector_type(4)))  int  v4i;

#define NH 12
#define DM 768
#define HD 64
#define SL 2048
#define NB 4
#define NROW (NB*SL)   // 8192

union FragB16 {
  v16bf v;
  uint4 q[2];
};

static __device__ __forceinline__ v8f wmma_bf16(v16bf a, v16bf b, v8f c) {
  // D = A(16x32) * B(32x16) + C, f32 accumulate
  return __builtin_amdgcn_wmma_f32_16x16x32_bf16(false, a, false, b, (short)0, c, false, false);
}

// ---------------------------------------------------------------------------
// Tensor Data Mover: 2D tile (64 x rows, 2-byte elems) global -> LDS.
// D# built per CDNA5 ISA ch.8: group0 = {flags, lds_addr, global_addr, type=2},
// group1 = {data_size=2B, dims/tile dims/stride}, groups 2/3 zero (2D tensor).
// This toolchain exposes the 6-arg builtin:
//   (uint32x4 g0, int32x8 g1, int32x4, int32x4, int32x8, i32 cpol)
// ---------------------------------------------------------------------------
#if defined(__AMDGCN__) && __has_builtin(__builtin_amdgcn_tensor_load_to_lds)
#define HAVE_TDM 1
static __device__ __forceinline__ void tdm_load_2d(const void* gaddr,
                                                   unsigned lds_byte_addr,
                                                   unsigned tile_d0,      // contig elems
                                                   unsigned tile_d1,      // rows
                                                   unsigned long long row_stride) // elems
{
  const unsigned long long ga = (unsigned long long)gaddr;
  v4u g0;
  g0[0] = 1u;                                            // count=1 (valid), no gather
  g0[1] = lds_byte_addr;                                 // lds_addr
  g0[2] = (unsigned)(ga & 0xFFFFFFFFu);                  // global_addr[31:0]
  g0[3] = (unsigned)((ga >> 32) & 0x1FFFFFFu) | (2u << 30); // addr[56:32] | type=2
  v8i g1;
  g1[0] = (int)(1u << 16);                               // data_size=1 (2 bytes)
  g1[1] = (int)((tile_d0 & 0xFFFFu) << 16);              // tensor_dim0[15:0]
  g1[2] = (int)(((tile_d0 >> 16) & 0xFFFFu) | ((tile_d1 & 0xFFFFu) << 16)); // d0 hi | d1 lo
  g1[3] = (int)(((tile_d1 >> 16) & 0xFFFFu) | ((tile_d0 & 0xFFFFu) << 16)); // d1 hi | tile_dim0
  g1[4] = (int)(tile_d1 & 0xFFFFu);                      // tile_dim1 (tile_dim2=0)
  g1[5] = (int)(row_stride & 0xFFFFFFFFu);               // tensor_dim0_stride lo
  g1[6] = (int)((row_stride >> 32) & 0xFFFFu);           // stride hi (dim1_stride unused)
  g1[7] = 0;
  const v4i z4 = {0, 0, 0, 0};                           // groups 2/3: 2D tensor
  const v8i z8 = {0, 0, 0, 0, 0, 0, 0, 0};               // extra group (clang-23 form)
  __builtin_amdgcn_tensor_load_to_lds(g0, g1, z4, z4, z8, 0);
}
#endif

static __device__ __forceinline__ void stage_tile_2d(const __bf16* __restrict__ g,
                                                     __bf16* lds, int row_stride_g,
                                                     int tid, int nthreads)
{
#ifdef HAVE_TDM
  if (tid < 32)
    tdm_load_2d(g, (unsigned)(unsigned long long)(void*)lds, 64u, 64u,
                (unsigned long long)row_stride_g);
  (void)nthreads;
#else
  for (int i = tid; i < 64 * 64 / 8; i += nthreads) {
    const int r = i >> 3, c = (i & 7) * 8;
    *(uint4*)&lds[r * 64 + c] = *(const uint4*)&g[(size_t)r * row_stride_g + c];
  }
#endif
}

static __device__ __forceinline__ void stage_fence(int tid) {
#ifdef HAVE_TDM
  if (tid < 32) {
#if __has_builtin(__builtin_amdgcn_s_wait_tensorcnt)
    __builtin_amdgcn_s_wait_tensorcnt((short)0);
#else
    asm volatile("s_wait_tensorcnt 0" ::: "memory");
#endif
  }
#endif
  __syncthreads();
}

// ---------------------------------------------------------------------------
// fp32 -> bf16 conversion
// ---------------------------------------------------------------------------
__global__ void cvt_f32_to_bf16(const float* __restrict__ src,
                                __bf16* __restrict__ dst, int n) {
  int i = blockIdx.x * blockDim.x + threadIdx.x;
  if (i < n) dst[i] = (__bf16)src[i];
}

// ---------------------------------------------------------------------------
// GEMM: out = X[NROW,DM] @ W^T + bias,  W is [DM(out_col), DM(k)] row-major.
// mode 0: Q  -> bf16 [B,H,L,HD], scaled by 1/8
// mode 1: K  -> bf16 [B,H,L,HD]
// mode 2: V  -> bf16 [B,H,HD,L]   (transposed for P@V B-fragments)
// mode 3: O  -> f32  [NROW, DM]   (final output)
// Block: 256 threads = 8 waves; block tile 128(M) x 64(N); wave tile 32x32.
// ---------------------------------------------------------------------------
__global__ __launch_bounds__(256)
void gemm_wmma(const __bf16* __restrict__ X,
               const __bf16* __restrict__ W,
               const float*  __restrict__ bias,
               void* __restrict__ out, int mode)
{
  const int lane = threadIdx.x & 31;
  const int wave = threadIdx.x >> 5;
  const int l15  = lane & 15;
  const int half = lane >> 4;

  const int m0 = blockIdx.x * 128 + (wave & 3) * 32;
  const int n0 = blockIdx.y * 64  + (wave >> 2) * 32;

  v8f acc[2][2];
  for (int a = 0; a < 2; a++)
    for (int bb = 0; bb < 2; bb++)
      for (int i = 0; i < 8; i++) acc[a][bb][i] = 0.f;

  const __bf16* xrow0 = X + (size_t)(m0 + l15)      * DM;
  const __bf16* xrow1 = X + (size_t)(m0 + 16 + l15) * DM;
  const __bf16* wrow0 = W + (size_t)(n0 + l15)      * DM;
  const __bf16* wrow1 = W + (size_t)(n0 + 16 + l15) * DM;

  for (int k0 = 0; k0 < DM; k0 += 32) {
    FragB16 a0, a1, b0, b1;
    const int ka = k0 + half * 8;       // A lane: 8 contig K + 8 contig K at +16
    a0.q[0] = *(const uint4*)(xrow0 + ka);
    a0.q[1] = *(const uint4*)(xrow0 + ka + 16);
    a1.q[0] = *(const uint4*)(xrow1 + ka);
    a1.q[1] = *(const uint4*)(xrow1 + ka + 16);
    const int kb = k0 + half * 8;       // B lane: 8 contig K + 8 contig K at +16
    b0.q[0] = *(const uint4*)(wrow0 + kb);
    b0.q[1] = *(const uint4*)(wrow0 + kb + 16);
    b1.q[0] = *(const uint4*)(wrow1 + kb);
    b1.q[1] = *(const uint4*)(wrow1 + kb + 16);

    if (k0 + 64 < DM) {                 // prefetch two k-blocks ahead
      __builtin_prefetch(xrow0 + ka + 64, 0, 0);
      __builtin_prefetch(xrow1 + ka + 64, 0, 0);
      __builtin_prefetch(wrow0 + kb + 64, 0, 0);
      __builtin_prefetch(wrow1 + kb + 64, 0, 0);
    }

    acc[0][0] = wmma_bf16(a0.v, b0.v, acc[0][0]);
    acc[0][1] = wmma_bf16(a0.v, b1.v, acc[0][1]);
    acc[1][0] = wmma_bf16(a1.v, b0.v, acc[1][0]);
    acc[1][1] = wmma_bf16(a1.v, b1.v, acc[1][1]);
  }

  for (int nt = 0; nt < 2; nt++) {
    const int n = n0 + nt * 16 + l15;
    const float bv = bias[n];
    for (int mt = 0; mt < 2; mt++) {
      for (int r = 0; r < 8; r++) {
        const int gm = m0 + mt * 16 + r + half * 8;  // C layout: M = r + 8*half
        float v = acc[mt][nt][r] + bv;
        if (mode == 3) {
          ((float*)out)[(size_t)gm * DM + n] = v;
        } else {
          const int bi = gm >> 11, lp = gm & (SL - 1);
          const int h = n >> 6, hd = n & 63;
          if (mode == 0) v *= 0.125f;  // 1/sqrt(HD) folded into Q
          const __bf16 bw = (__bf16)v;
          if (mode == 2) {
            ((__bf16*)out)[(((size_t)(bi * NH + h)) * HD + hd) * SL + lp] = bw;
          } else {
            ((__bf16*)out)[(((size_t)(bi * NH + h)) * SL + lp) * HD + hd] = bw;
          }
        }
      }
    }
  }
}

// ---------------------------------------------------------------------------
// Causal flash attention. Grid (L/64, H, B); 128 threads = 4 waves.
// K / V^T tiles staged to LDS by the Tensor Data Mover each 64-key block.
// Wave owns 16 query rows; block-uniform causal loop bound.
// ---------------------------------------------------------------------------
__global__ __launch_bounds__(128)
void flash_attn(const __bf16* __restrict__ Q,   // [B,H,L,HD] (pre-scaled)
                const __bf16* __restrict__ K,   // [B,H,L,HD]
                const __bf16* __restrict__ Vt,  // [B,H,HD,L]
                __bf16* __restrict__ Aout)      // [NROW, DM]
{
  __shared__ __align__(16) __bf16 Kt[64][64];       // keys x hd      (8 KB)
  __shared__ __align__(16) __bf16 Vs[64][64];       // hd x keys      (8 KB)
  __shared__ __align__(16) __bf16 Plds[4][16][64];  // per-wave P     (8 KB)

  const int tid  = threadIdx.x;
  const int lane = tid & 31;
  const int wave = tid >> 5;
  const int l15  = lane & 15;
  const int half = lane >> 4;
  const int h  = blockIdx.y;
  const int b  = blockIdx.z;
  const int q0 = blockIdx.x * 64 + wave * 16;

  const size_t bh = (size_t)(b * NH + h);
  const __bf16* Qp = Q  + bh * SL * HD;
  const __bf16* Kp = K  + bh * SL * HD;
  const __bf16* Vp = Vt + bh * HD * SL;

  // Q A-fragments: 2 chunks along HD (k=0..31, 32..63), fixed for whole loop
  FragB16 qf[2];
  {
    const __bf16* qrow = Qp + (size_t)(q0 + l15) * HD;
    for (int c = 0; c < 2; c++) {
      const int kb = c * 32 + half * 8;
      qf[c].q[0] = *(const uint4*)(qrow + kb);
      qf[c].q[1] = *(const uint4*)(qrow + kb + 16);
    }
  }

  v8f o[4];
  for (int t = 0; t < 4; t++)
    for (int i = 0; i < 8; i++) o[t][i] = 0.f;
  float mrow[8], lrow[8];
  for (int r = 0; r < 8; r++) { mrow[r] = -1e30f; lrow[r] = 0.f; }

  const int kbMaxB = blockIdx.x;  // block-uniform causal bound (64-row tile)
  for (int kb = 0; kb <= kbMaxB; kb++) {
    const int kbase = kb * 64;

    // ---- TDM-stage K tile [64 keys x 64 hd] and V^T tile [64 hd x 64 keys] ----
    stage_tile_2d(Kp + (size_t)kbase * HD, &Kt[0][0], HD, tid, 128);
    stage_tile_2d(Vp + kbase,              &Vs[0][0], SL, tid, 128);
    stage_fence(tid);

    // ---- S = Q @ K^T for 4 key subtiles of 16 ----
    v8f s[4];
    for (int t = 0; t < 4; t++) {
      FragB16 kf0, kf1;  // B frags from LDS: lane = key column, K-dim 8 contig + 8 contig at +16
      const __bf16* krow = &Kt[t * 16 + l15][half * 8];
      kf0.q[0] = *(const uint4*)(krow);
      kf0.q[1] = *(const uint4*)(krow + 16);
      kf1.q[0] = *(const uint4*)(krow + 32);
      kf1.q[1] = *(const uint4*)(krow + 48);
      v8f c;
      for (int i = 0; i < 8; i++) c[i] = 0.f;
      c = wmma_bf16(qf[0].v, kf0.v, c);
      c = wmma_bf16(qf[1].v, kf1.v, c);
      s[t] = c;
    }

    // ---- causal mask (branchless) ----
    for (int t = 0; t < 4; t++) {
      const int key = kbase + t * 16 + l15;
      for (int r = 0; r < 8; r++) {
        const int row = q0 + r + half * 8;
        s[t][r] = (key > row) ? -1e30f : s[t][r];
      }
    }

    // ---- online softmax: row stats across 16 lanes of each half-wave ----
    for (int r = 0; r < 8; r++) {
      float mx = fmaxf(fmaxf(s[0][r], s[1][r]), fmaxf(s[2][r], s[3][r]));
      for (int off = 8; off; off >>= 1) mx = fmaxf(mx, __shfl_xor(mx, off, 32));
      const float mnew = fmaxf(mrow[r], mx);
      const float sc = __expf(mrow[r] - mnew);
      mrow[r] = mnew;
      float ps = 0.f;
      for (int t = 0; t < 4; t++) {
        const float p = __expf(s[t][r] - mnew);
        s[t][r] = p;
        ps += p;
      }
      for (int off = 8; off; off >>= 1) ps += __shfl_xor(ps, off, 32);
      lrow[r] = lrow[r] * sc + ps;
      for (int t = 0; t < 4; t++) o[t][r] *= sc;
    }

    // ---- P: C-layout regs -> LDS -> A-layout fragments ----
    for (int t = 0; t < 4; t++)
      for (int r = 0; r < 8; r++)
        Plds[wave][r + half * 8][t * 16 + l15] = (__bf16)s[t][r];
    asm volatile("s_wait_dscnt 0" ::: "memory");

    // ---- O += P @ V (V^T tile in LDS: contiguous B-fragments) ----
    for (int c = 0; c < 2; c++) {
      FragB16 pf;
      const __bf16* prow = &Plds[wave][l15][0];
      const int ko = c * 32 + half * 8;
      pf.q[0] = *(const uint4*)(prow + ko);
      pf.q[1] = *(const uint4*)(prow + ko + 16);
      for (int t = 0; t < 4; t++) {
        FragB16 vf;  // lane = hd column, K-dim = 8 contig keys + 8 contig keys at +16
        const __bf16* vrow = &Vs[t * 16 + l15][c * 32 + half * 8];
        vf.q[0] = *(const uint4*)(vrow);
        vf.q[1] = *(const uint4*)(vrow + 16);
        o[t] = wmma_bf16(pf.v, vf.v, o[t]);
      }
    }
    __syncthreads();  // protect Kt/Vs before next TDM staging
  }

  // ---- epilogue: normalize and store to [NROW, DM] ----
  float inv[8];
  for (int r = 0; r < 8; r++) inv[r] = 1.f / lrow[r];
  for (int t = 0; t < 4; t++) {
    for (int r = 0; r < 8; r++) {
      const int row = q0 + r + half * 8;
      const size_t gm = (size_t)b * SL + row;
      const int n = h * 64 + t * 16 + l15;
      Aout[gm * DM + n] = (__bf16)(o[t][r] * inv[r]);
    }
  }
}

// ---------------------------------------------------------------------------
extern "C" void kernel_launch(void* const* d_in, const int* in_sizes, int n_in,
                              void* d_out, int out_size, void* d_ws, size_t ws_size,
                              hipStream_t stream) {
  const float* hs = (const float*)d_in[0];
  const float* wq = (const float*)d_in[1];
  const float* bq = (const float*)d_in[2];
  const float* wk = (const float*)d_in[3];
  const float* bk = (const float*)d_in[4];
  const float* wv = (const float*)d_in[5];
  const float* bv = (const float*)d_in[6];
  const float* wo = (const float*)d_in[7];
  const float* bo = (const float*)d_in[8];

  char* ws = (char*)d_ws;
  size_t off = 0;
  auto alloc = [&](size_t bytes) {
    void* p = ws + off;
    off = (off + bytes + 255) & ~(size_t)255;
    return p;
  };
  const size_t XN = (size_t)NROW * DM;      // 6291456 elems
  const size_t WN = (size_t)DM * DM;        // 589824 elems
  __bf16* Xb  = (__bf16*)alloc(XN * 2);
  __bf16* Wqb = (__bf16*)alloc(WN * 2);
  __bf16* Wkb = (__bf16*)alloc(WN * 2);
  __bf16* Wvb = (__bf16*)alloc(WN * 2);
  __bf16* Wob = (__bf16*)alloc(WN * 2);
  __bf16* Qb  = (__bf16*)alloc(XN * 2);     // [B,H,L,HD]
  __bf16* Kb  = (__bf16*)alloc(XN * 2);     // [B,H,L,HD]
  __bf16* Vtb = (__bf16*)alloc(XN * 2);     // [B,H,HD,L]
  __bf16* Ab  = (__bf16*)alloc(XN * 2);     // attention out [NROW, DM]

  cvt_f32_to_bf16<<<(int)((XN + 255) / 256), 256, 0, stream>>>(hs, Xb, (int)XN);
  cvt_f32_to_bf16<<<(int)((WN + 255) / 256), 256, 0, stream>>>(wq, Wqb, (int)WN);
  cvt_f32_to_bf16<<<(int)((WN + 255) / 256), 256, 0, stream>>>(wk, Wkb, (int)WN);
  cvt_f32_to_bf16<<<(int)((WN + 255) / 256), 256, 0, stream>>>(wv, Wvb, (int)WN);
  cvt_f32_to_bf16<<<(int)((WN + 255) / 256), 256, 0, stream>>>(wo, Wob, (int)WN);

  dim3 ggrid(NROW / 128, DM / 64);
  gemm_wmma<<<ggrid, 256, 0, stream>>>(Xb, Wqb, bq, Qb,  0);
  gemm_wmma<<<ggrid, 256, 0, stream>>>(Xb, Wkb, bk, Kb,  1);
  gemm_wmma<<<ggrid, 256, 0, stream>>>(Xb, Wvb, bv, Vtb, 2);

  flash_attn<<<dim3(SL / 64, NH, NB), 128, 0, stream>>>(Qb, Kb, Vtb, Ab);

  gemm_wmma<<<ggrid, 256, 0, stream>>>(Ab, Wob, bo, d_out, 3);
}
